// CSOCSSC_v46_66949950210598
// MI455X (gfx1250) — matrix operand order repaired and checked by reading, not compiled, not run on hardware
//
#include <hip/hip_runtime.h>
#include <hip/hip_bf16.h>
#include <cstdint>
#include <cstddef>

#define N_NODES  32768
#define N_EDGES  524288
#define NODE_DIM 128
#define HIDDEN   256
#define EDGE_DIM 32
#define MIN_DIM  288   // 2*NODE_DIM + EDGE_DIM

typedef __attribute__((ext_vector_type(16))) __bf16        v16bf;
typedef __attribute__((ext_vector_type(8)))  float         v8f;
typedef __attribute__((ext_vector_type(4)))  unsigned int  u32x4;

union Frag16 { v16bf v; u32x4 q[2]; };

__device__ __forceinline__ unsigned short f2bf(float f) {
  unsigned int u = __float_as_uint(f);
  u += 0x7fffu + ((u >> 16) & 1u);         // round-to-nearest-even
  return (unsigned short)(u >> 16);
}

__device__ __forceinline__ float silu_f(float x) {
  return x / (1.0f + __expf(-x));
}

// ---------------------------------------------------------------------------
// Pack a row-major f32 weight [Ktot x Ntot] into bf16 WMMA B-fragment order.
// Fragment f = (ntile * nK + kstep), 512 halves each.
// Within a fragment: half index = lane*16 + j, value = W[k][n] with
//   k = kstep*32 + (lane<16 ? 0 : 8) + (j<8 ? j : j+8),   n = ntile*16 + (lane & 15)
// ---------------------------------------------------------------------------
__global__ void pack_w_kernel(const float* __restrict__ W,
                              unsigned short* __restrict__ out,
                              int Ktot, int Ntot) {
  int nK = Ktot >> 5;
  int total = Ktot * Ntot;
  for (int idx = blockIdx.x * blockDim.x + threadIdx.x; idx < total;
       idx += gridDim.x * blockDim.x) {
    int f = idx >> 9, r = idx & 511;
    int l = r >> 4, j = r & 15;
    int t = f / nK, s = f - t * nK;
    int k = s * 32 + ((l < 16) ? 0 : 8) + ((j < 8) ? j : j + 8);
    int n = t * 16 + (l & 15);
    out[idx] = f2bf(W[k * Ntot + n]);
  }
}

// Initialize d_out = [h ; x] so scatter-adds compute h+h_agg, x+x_agg.
__global__ void init_out_kernel(const float* __restrict__ h,
                                const float* __restrict__ x,
                                float* __restrict__ out) {
  const int nh = N_NODES * NODE_DIM;
  const int nx = N_NODES * 3;
  int i = blockIdx.x * blockDim.x + threadIdx.x;
  if (i < nh)            out[i] = h[i];
  else if (i < nh + nx)  out[i] = x[i - nh];
}

// ---------------------------------------------------------------------------
// Fused EGNN edge kernel: 64 edges per block, 256 threads (8 wave32).
// ---------------------------------------------------------------------------
__global__ __launch_bounds__(256)
void egnn_edge_kernel(const float* __restrict__ hg, const float* __restrict__ xg,
                      const long long* __restrict__ ei, const float* __restrict__ dist,
                      const float* __restrict__ We1, const float* __restrict__ be1,
                      const float* __restrict__ We2, const float* __restrict__ be2,
                      const float* __restrict__ bn1, const float* __restrict__ bn2,
                      const float* __restrict__ bc1, const float* __restrict__ Wc2,
                      const unsigned short* __restrict__ pWn1,
                      const unsigned short* __restrict__ pWc1,
                      const unsigned short* __restrict__ pWn2,
                      float* __restrict__ outH, float* __restrict__ outX) {
  constexpr int EPB = 64;    // edges per block
  constexpr int MST = 296;   // m_input LDS row stride (halves, 16B-aligned rows)
  constexpr int HST = 264;   // hidden  LDS row stride (halves, 16B-aligned rows)

  __shared__ __align__(16) unsigned short lds_buf[EPB * MST]; // m_input, reused for hidden
  __shared__ float lds_silu[EPB * EDGE_DIM];
  __shared__ float lds_cw[EPB];
  __shared__ int   lds_src[EPB];
  __shared__ int   lds_dst[EPB];

  const int tid  = threadIdx.x;
  const int lane = tid & 31;
  const int w    = tid >> 5;        // wave id 0..7
  const int eb   = blockIdx.x * EPB;

  // ---- P0: edge indices + zero coord accumulator -------------------------
  if (tid < EPB) {
    lds_src[tid] = (int)ei[eb + tid];
    lds_dst[tid] = (int)ei[N_EDGES + eb + tid];
    lds_cw[tid]  = 0.0f;
  }
  __syncthreads();

  // ---- P1a: edge MLP first layer silu values -----------------------------
  for (int idx = tid; idx < EPB * EDGE_DIM; idx += 256) {
    int e = idx >> 5, k = idx & 31;
    float d = dist[eb + e];
    lds_silu[idx] = silu_f(d * We1[k] + be1[k]);
  }

  // ---- P1b: gather h[src], h[dst] rows into LDS as bf16 ------------------
  {
    int e = tid >> 2, sub = tid & 3;          // 4 threads per edge
    int s = lds_src[e], d = lds_dst[e];
    const float4* srow = (const float4*)(hg + s * NODE_DIM);
    const float4* drow = (const float4*)(hg + d * NODE_DIM);
    unsigned short* mrow = lds_buf + e * MST;
#pragma unroll
    for (int c4 = 0; c4 < 16; ++c4) {
      int cc4 = sub * 16 + c4;                // float4 index 0..63 across 256 cols
      float4 v = (cc4 < 32) ? srow[cc4] : drow[cc4 - 32];
      int c = cc4 * 4;
      mrow[c + 0] = f2bf(v.x); mrow[c + 1] = f2bf(v.y);
      mrow[c + 2] = f2bf(v.z); mrow[c + 3] = f2bf(v.w);
    }
  }
  __syncthreads();

  // ---- P1c: edge MLP second layer -> m_input cols [256,288) --------------
  for (int idx = tid; idx < EPB * EDGE_DIM; idx += 256) {
    int e = idx >> 5, o = idx & 31;
    float acc = be2[o];
#pragma unroll
    for (int k = 0; k < 32; ++k) acc += lds_silu[e * 32 + k] * We2[k * 32 + o];
    lds_buf[e * MST + 2 * NODE_DIM + o] = f2bf(acc);
  }
  __syncthreads();

  // A fragment: per-lane K chunks {ks..ks+7, ks+16..ks+23}, ks=32s+(lane<16?0:8)
  auto loadA = [&](const unsigned short* base, int stride, int mt, int s) -> Frag16 {
    int row = mt * 16 + (lane & 15);
    int ks  = s * 32 + ((lane < 16) ? 0 : 8);
    Frag16 f;
    f.q[0] = *(const u32x4*)(base + row * stride + ks);
    f.q[1] = *(const u32x4*)(base + row * stride + ks + 16);
    return f;
  };
  // B fragment: packed, 512 halves per fragment, 32B contiguous per lane
  auto loadB = [&](const unsigned short* pw, int nK, int t, int s) -> Frag16 {
    const u32x4* p = (const u32x4*)(pw + ((t * nK + s) * 512 + lane * 16));
    Frag16 f; f.q[0] = p[0]; f.q[1] = p[1]; return f;
  };

  const v8f vzero = {};

  // ---- P2a: coord path L1 (fold directly into per-edge scalar) -----------
#pragma unroll
  for (int t2 = 0; t2 < 2; ++t2) {
    int t = 2 * w + t2;                       // N-tile of HIDDEN (16 tiles / 8 waves)
#pragma unroll
    for (int mt = 0; mt < 4; ++mt) {
      v8f acc = vzero;
      for (int s = 0; s < 9; ++s) {
        Frag16 a = loadA(lds_buf, MST, mt, s);
        Frag16 b = loadB(pWc1, 9, t, s);
        acc = __builtin_amdgcn_wmma_f32_16x16x32_bf16(
            false, a.v, false, b.v, (short)0, acc, false, false);
      }
      int n = t * 16 + (lane & 15);
      float wv = Wc2[n];
      float bb = bc1[n];
      int rbase = mt * 16 + ((lane < 16) ? 0 : 8);
#pragma unroll
      for (int r = 0; r < 8; ++r)
        atomicAdd(&lds_cw[rbase + r], silu_f(acc[r] + bb) * wv);
    }
  }

  // ---- P2b: node path L1, accumulators in registers ----------------------
  v8f hacc[2][4];
#pragma unroll
  for (int t2 = 0; t2 < 2; ++t2)
#pragma unroll
    for (int mt = 0; mt < 4; ++mt) hacc[t2][mt] = vzero;

#pragma unroll
  for (int t2 = 0; t2 < 2; ++t2) {
    int t = 2 * w + t2;
#pragma unroll
    for (int mt = 0; mt < 4; ++mt) {
      for (int s = 0; s < 9; ++s) {
        Frag16 a = loadA(lds_buf, MST, mt, s);
        Frag16 b = loadB(pWn1, 9, t, s);
        hacc[t2][mt] = __builtin_amdgcn_wmma_f32_16x16x32_bf16(
            false, a.v, false, b.v, (short)0, hacc[t2][mt], false, false);
      }
    }
  }
  __syncthreads();   // all reads of m_input (and lds_cw adds) done

  // ---- write silu(hidden) as bf16 over the m_input region ----------------
#pragma unroll
  for (int t2 = 0; t2 < 2; ++t2) {
    int n = (2 * w + t2) * 16 + (lane & 15);
    float bb = bn1[n];
#pragma unroll
    for (int mt = 0; mt < 4; ++mt) {
      int rbase = mt * 16 + ((lane < 16) ? 0 : 8);
#pragma unroll
      for (int r = 0; r < 8; ++r)
        lds_buf[(rbase + r) * HST + n] = f2bf(silu_f(hacc[t2][mt][r] + bb));
    }
  }
  __syncthreads();

  // ---- P3: node path L2 (K=256) + scatter-add into h output --------------
  {
    v8f macc[4];
#pragma unroll
    for (int mt = 0; mt < 4; ++mt) macc[mt] = vzero;
#pragma unroll
    for (int mt = 0; mt < 4; ++mt) {
      for (int s = 0; s < 8; ++s) {
        Frag16 a = loadA(lds_buf, HST, mt, s);
        Frag16 b = loadB(pWn2, 8, w, s);      // wave w owns output cols [16w,16w+16)
        macc[mt] = __builtin_amdgcn_wmma_f32_16x16x32_bf16(
            false, a.v, false, b.v, (short)0, macc[mt], false, false);
      }
    }
    int col = w * 16 + (lane & 15);
    float bb = bn2[col];
#pragma unroll
    for (int mt = 0; mt < 4; ++mt) {
      int rbase = mt * 16 + ((lane < 16) ? 0 : 8);
#pragma unroll
      for (int r = 0; r < 8; ++r) {
        int row = rbase + r;
        int d = lds_dst[row];
        atomicAdd(outH + d * NODE_DIM + col, macc[mt][r] + bb);
      }
    }
  }

  // ---- P4: equivariant coordinate update ---------------------------------
  if (tid < EPB) {
    int e = tid;
    float cw = lds_cw[e];
    int s = lds_src[e], d = lds_dst[e];
    float dx = xg[s * 3 + 0] - xg[d * 3 + 0];
    float dy = xg[s * 3 + 1] - xg[d * 3 + 1];
    float dz = xg[s * 3 + 2] - xg[d * 3 + 2];
    float len = fmaxf(sqrtf(dx * dx + dy * dy + dz * dz), 1e-8f);
    float sc = cw / len;
    atomicAdd(outX + d * 3 + 0, sc * dx);
    atomicAdd(outX + d * 3 + 1, sc * dy);
    atomicAdd(outX + d * 3 + 2, sc * dz);
  }
}

// ---------------------------------------------------------------------------
extern "C" void kernel_launch(void* const* d_in, const int* in_sizes, int n_in,
                              void* d_out, int out_size, void* d_ws, size_t ws_size,
                              hipStream_t stream) {
  const float*     h    = (const float*)d_in[0];
  const float*     x    = (const float*)d_in[1];
  const long long* ei   = (const long long*)d_in[2];
  const float*     dist = (const float*)d_in[3];
  const float*     We1  = (const float*)d_in[4];
  const float*     be1  = (const float*)d_in[5];
  const float*     We2  = (const float*)d_in[6];
  const float*     be2  = (const float*)d_in[7];
  const float*     Wn1  = (const float*)d_in[8];
  const float*     bn1  = (const float*)d_in[9];
  const float*     Wn2  = (const float*)d_in[10];
  const float*     bn2  = (const float*)d_in[11];
  const float*     Wc1  = (const float*)d_in[12];
  const float*     bc1  = (const float*)d_in[13];
  const float*     Wc2  = (const float*)d_in[14];

  unsigned short* ws   = (unsigned short*)d_ws;
  unsigned short* pWn1 = ws;                           // 288*256 halves
  unsigned short* pWc1 = ws + MIN_DIM * HIDDEN;        // 288*256
  unsigned short* pWn2 = ws + 2 * MIN_DIM * HIDDEN;    // 256*128

  pack_w_kernel<<<288, 256, 0, stream>>>(Wn1, pWn1, MIN_DIM, HIDDEN);
  pack_w_kernel<<<288, 256, 0, stream>>>(Wc1, pWc1, MIN_DIM, HIDDEN);
  pack_w_kernel<<<128, 256, 0, stream>>>(Wn2, pWn2, HIDDEN, NODE_DIM);

  int tot = N_NODES * NODE_DIM + N_NODES * 3;
  init_out_kernel<<<(tot + 255) / 256, 256, 0, stream>>>(h, x, (float*)d_out);

  float* outH = (float*)d_out;
  float* outX = outH + (size_t)N_NODES * NODE_DIM;
  egnn_edge_kernel<<<N_EDGES / 64, 256, 0, stream>>>(
      h, x, ei, dist, We1, be1, We2, be2, bn1, bn2, bc1, Wc2,
      pWn1, pWc1, pWn2, outH, outX);
}
